// LSTMSimpleModel_85899346555
// MI455X (gfx1250) — matrix operand order repaired and checked by reading, not compiled, not run on hardware
//
#include <hip/hip_runtime.h>

// ---- problem sizes (match reference) ----
#define I_SZ 256
#define H_SZ 1024
#define B_SZ 64
#define T_SZ 256
#define C_SZ 60
#define KC   (I_SZ + H_SZ)   // 1280 combined K = [x | h]
#define G4   (4 * H_SZ)      // 4096 gate rows
#define NBLK 64              // persistent blocks (16 h-cols each)

typedef __attribute__((ext_vector_type(16))) __bf16 v16bf;
typedef __attribute__((ext_vector_type(8)))  float  v8f;

union V16 { v16bf v; uint4 q[2]; };

static __device__ __forceinline__ unsigned short f2bf(float f) {
  unsigned int u = __float_as_uint(f);
  u += 0x7FFFu + ((u >> 16) & 1u);          // round-to-nearest-even
  return (unsigned short)(u >> 16);
}
static __device__ __forceinline__ float sigf(float x) {
  return 1.0f / (1.0f + __expf(-x));
}

// ---- prep: inputs [B,T,I] f32 -> x_bf [T,B,I] bf16 ----
__global__ void prep_x(const float* __restrict__ in, unsigned short* __restrict__ xbf) {
  int idx = blockIdx.x * blockDim.x + threadIdx.x;
  if (idx >= B_SZ * T_SZ * I_SZ) return;
  int i = idx % I_SZ;
  int t = (idx / I_SZ) % T_SZ;
  int b = idx / (I_SZ * T_SZ);
  xbf[(size_t)t * B_SZ * I_SZ + b * I_SZ + i] = f2bf(in[idx]);
}

// ---- prep: pack Wc[n,k] = k<I ? W_ih[n,k] : W_hh[n,k-I], bf16 ----
__global__ void prep_w(const float* __restrict__ Wih, const float* __restrict__ Whh,
                       unsigned short* __restrict__ Wc) {
  int idx = blockIdx.x * blockDim.x + threadIdx.x;
  if (idx >= G4 * KC) return;
  int k = idx % KC;
  int n = idx / KC;
  float v = (k < I_SZ) ? Wih[n * I_SZ + k] : Whh[n * H_SZ + (k - I_SZ)];
  Wc[idx] = f2bf(v);
}

// ---- prep: combined bias; zero h buffers and barrier ----
__global__ void prep_misc(const float* __restrict__ bih, const float* __restrict__ bhh,
                          float* __restrict__ bc, unsigned short* __restrict__ h0,
                          unsigned short* __restrict__ h1, unsigned* __restrict__ bar) {
  int idx = blockIdx.x * blockDim.x + threadIdx.x;
  if (idx < G4) bc[idx] = bih[idx] + bhh[idx];
  if (idx < B_SZ * H_SZ) { h0[idx] = 0; h1[idx] = 0; }
  if (idx == 0) *bar = 0u;
}

// ---- persistent LSTM: all T steps, device-wide barrier between steps ----
// grid = 64 blocks (16 h-cols each) x 64 threads (2 waves, 2 M-tiles each).
// Cell state c lives entirely in registers for all 256 steps.
__global__ __launch_bounds__(64) void lstm_persist(
    const unsigned short* __restrict__ xbf,  // [T, B, I] bf16
    unsigned short* __restrict__ h0,         // [B, H] bf16 (read at even t)
    unsigned short* __restrict__ h1,         // [B, H] bf16 (read at odd t)
    const unsigned short* __restrict__ Wc,   // [4H, KC] bf16
    const float* __restrict__ bc,            // [4H]
    float* __restrict__ hf,                  // [B, H] f32 (last h, for FC)
    unsigned* __restrict__ bar)              // monotone arrival counter
{
  const int lane  = threadIdx.x & 31;
  const int wv    = threadIdx.x >> 5;        // 0..1 -> M-tile pair
  const int n0    = blockIdx.x * 16;         // h-column base
  const int row16 = lane & 15;
  const int khalf = lane >> 4;               // 0 or 1
  const int mA0   = wv * 32;                 // M-tiles at mA0 and mA0+16

  // biases for this lane's column (constant across all steps)
  const int n = n0 + row16;
  const float bi = bc[n];
  const float bf_ = bc[H_SZ + n];
  const float bg = bc[2 * H_SZ + n];
  const float bo = bc[3 * H_SZ + n];

  // per-lane weight row offsets (gate g, column n): constant across steps
  size_t woff[4];
  #pragma unroll
  for (int g = 0; g < 4; ++g)
    woff[g] = (size_t)(g * H_SZ + n) * KC + khalf * 8;

  float creg[16] = {};                       // cell state: [tile 0..1][j 0..7]

  for (int t = 0; t < T_SZ; ++t) {
    // Launder a ZERO OFFSET (not the pointer) each step: the volatile asm
    // makes weight addresses loop-variant so LICM cannot hoist/spill the
    // tiles across the t-loop, while the base pointer keeps its global
    // address-space provenance -> global_load_b128 (not flat_load).
    size_t zoff = 0;
    asm volatile("" : "+s"(zoff));
    const unsigned short* wcl = Wc + zoff;

    const unsigned short* xt = xbf + (size_t)t * B_SZ * I_SZ;
    v8f acc[8] = {};                         // [tile 0..1][gate i,f,g,o]

    // ---- x-projection part (K = 0..255): independent of h -> before barrier
    #pragma unroll 2
    for (int k0 = 0; k0 < I_SZ; k0 += 32) {
      V16 a0, a1;
      const unsigned short* ar0 = xt + (size_t)(mA0 + row16) * I_SZ + k0;
      const unsigned short* ar1 = ar0 + (size_t)16 * I_SZ;
      a0.q[0] = *(const uint4*)(ar0 + khalf * 8);
      a0.q[1] = *(const uint4*)(ar0 + 16 + khalf * 8);
      a1.q[0] = *(const uint4*)(ar1 + khalf * 8);
      a1.q[1] = *(const uint4*)(ar1 + 16 + khalf * 8);
      #pragma unroll
      for (int g = 0; g < 4; ++g) {
        V16 b;
        const unsigned short* wr = wcl + woff[g] + k0;
        b.q[0] = *(const uint4*)(wr);
        b.q[1] = *(const uint4*)(wr + 16);
        acc[g]     = __builtin_amdgcn_wmma_f32_16x16x32_bf16(
                       false, a0.v, false, b.v, (short)0, acc[g], false, false);
        acc[4 + g] = __builtin_amdgcn_wmma_f32_16x16x32_bf16(
                       false, a1.v, false, b.v, (short)0, acc[4 + g], false, false);
      }
    }

    // ---- wait for h_t from all blocks (skip at t=0: h0 pre-zeroed)
    if (t > 0) {
      if (threadIdx.x == 0) {
        while (__hip_atomic_load(bar, __ATOMIC_RELAXED, __HIP_MEMORY_SCOPE_AGENT)
               < (unsigned)(NBLK * t))
          __builtin_amdgcn_s_sleep(1);
      }
      __syncthreads();
      __builtin_amdgcn_fence(__ATOMIC_ACQUIRE, "agent");   // global_inv
    }
    const unsigned short* hin = (t & 1) ? h1 : h0;
    unsigned short* hout      = (t & 1) ? h0 : h1;

    // ---- recurrent part (K = 256..1279)
    #pragma unroll 2
    for (int k0 = 0; k0 < H_SZ; k0 += 32) {
      V16 a0, a1;
      const unsigned short* ar0 = hin + (size_t)(mA0 + row16) * H_SZ + k0;
      const unsigned short* ar1 = ar0 + (size_t)16 * H_SZ;
      a0.q[0] = *(const uint4*)(ar0 + khalf * 8);
      a0.q[1] = *(const uint4*)(ar0 + 16 + khalf * 8);
      a1.q[0] = *(const uint4*)(ar1 + khalf * 8);
      a1.q[1] = *(const uint4*)(ar1 + 16 + khalf * 8);
      #pragma unroll
      for (int g = 0; g < 4; ++g) {
        V16 b;
        const unsigned short* wr = wcl + woff[g] + I_SZ + k0;
        b.q[0] = *(const uint4*)(wr);
        b.q[1] = *(const uint4*)(wr + 16);
        acc[g]     = __builtin_amdgcn_wmma_f32_16x16x32_bf16(
                       false, a0.v, false, b.v, (short)0, acc[g], false, false);
        acc[4 + g] = __builtin_amdgcn_wmma_f32_16x16x32_bf16(
                       false, a1.v, false, b.v, (short)0, acc[4 + g], false, false);
      }
    }

    // ---- gates + cell update; C/D layout: VGPR j -> M = tile*16 + khalf*8 + j
    #pragma unroll
    for (int tile = 0; tile < 2; ++tile) {
      const int mbase = mA0 + tile * 16 + khalf * 8;
      #pragma unroll
      for (int j = 0; j < 8; ++j) {
        const int m = mbase + j;
        const size_t p = (size_t)m * H_SZ + n;
        float iv = sigf(acc[tile * 4 + 0][j] + bi);
        float fv = sigf(acc[tile * 4 + 1][j] + bf_);
        float gv = tanhf(acc[tile * 4 + 2][j] + bg);
        float ov = sigf(acc[tile * 4 + 3][j] + bo);
        float cn = fv * creg[tile * 8 + j] + iv * gv;
        float hn = ov * tanhf(cn);
        creg[tile * 8 + j] = cn;
        hout[p] = f2bf(hn);
        if (t == T_SZ - 1) hf[p] = hn;
      }
    }

    // ---- publish h_{t+1}: release stores, then arrive
    __builtin_amdgcn_fence(__ATOMIC_RELEASE, "agent");     // global_wb
    __syncthreads();
    if (threadIdx.x == 0)
      __hip_atomic_fetch_add(bar, 1u, __ATOMIC_RELAXED, __HIP_MEMORY_SCOPE_AGENT);
  }
}

// ---- final FC + log_softmax: one block per batch row ----
__global__ void fc_logsoftmax(const float* __restrict__ h, const float* __restrict__ fw,
                              const float* __restrict__ fb, float* __restrict__ out) {
  __shared__ float sl[C_SZ];
  __shared__ float red[2];
  const int b  = blockIdx.x;
  const int cc = threadIdx.x;
  if (cc < C_SZ) {
    float s = fb[cc];
    const float* hr = h + (size_t)b * H_SZ;
    const float* wr = fw + (size_t)cc * H_SZ;
    for (int k = 0; k < H_SZ; ++k) s += hr[k] * wr[k];
    sl[cc] = s;
  }
  __syncthreads();
  if (threadIdx.x == 0) {
    float mx = sl[0];
    for (int k = 1; k < C_SZ; ++k) mx = fmaxf(mx, sl[k]);
    float sum = 0.0f;
    for (int k = 0; k < C_SZ; ++k) sum += __expf(sl[k] - mx);
    red[0] = mx;
    red[1] = __logf(sum);
  }
  __syncthreads();
  if (cc < C_SZ) out[(size_t)b * C_SZ + cc] = sl[cc] - red[0] - red[1];
}

extern "C" void kernel_launch(void* const* d_in, const int* in_sizes, int n_in,
                              void* d_out, int out_size, void* d_ws, size_t ws_size,
                              hipStream_t stream) {
  const float* inputs = (const float*)d_in[0];
  const float* W_ih   = (const float*)d_in[1];
  const float* W_hh   = (const float*)d_in[2];
  const float* b_ih   = (const float*)d_in[3];
  const float* b_hh   = (const float*)d_in[4];
  const float* fc_w   = (const float*)d_in[5];
  const float* fc_b   = (const float*)d_in[6];
  float* out = (float*)d_out;

  char* ws = (char*)d_ws;
  size_t off = 0;
  auto alloc = [&](size_t bytes) -> void* {
    void* p = ws + off;
    off = (off + bytes + 255) & ~(size_t)255;
    return p;
  };
  unsigned short* xbf = (unsigned short*)alloc((size_t)T_SZ * B_SZ * I_SZ * 2); //  8 MB
  unsigned short* Wc  = (unsigned short*)alloc((size_t)G4 * KC * 2);            // 10 MB
  float*          bcv = (float*)alloc((size_t)G4 * 4);
  unsigned short* h0  = (unsigned short*)alloc((size_t)B_SZ * H_SZ * 2);
  unsigned short* h1  = (unsigned short*)alloc((size_t)B_SZ * H_SZ * 2);
  float*          hf  = (float*)alloc((size_t)B_SZ * H_SZ * 4);
  unsigned*       bar = (unsigned*)alloc(256);

  prep_x<<<(B_SZ * T_SZ * I_SZ + 255) / 256, 256, 0, stream>>>(inputs, xbf);
  prep_w<<<(G4 * KC + 255) / 256, 256, 0, stream>>>(W_ih, W_hh, Wc);
  prep_misc<<<(B_SZ * H_SZ + 255) / 256, 256, 0, stream>>>(b_ih, b_hh, bcv, h0, h1, bar);

  lstm_persist<<<NBLK, 64, 0, stream>>>(xbf, h0, h1, Wc, bcv, hf, bar);

  fc_logsoftmax<<<64, 64, 0, stream>>>(hf, fc_w, fc_b, out);
}
